// RetinaNetP45P6_58815282152374
// MI455X (gfx1250) — matrix operand order repaired and checked by reading, not compiled, not run on hardware
//
#include <hip/hip_runtime.h>
#include <hip/hip_bf16.h>
#include <math.h>

// ---------------------------------------------------------------------------
// Types for CDNA5 WMMA (wave32, 16x16x32 bf16 -> f32)
// ---------------------------------------------------------------------------
typedef __bf16 bf16_t;
typedef __attribute__((ext_vector_type(16))) __bf16 v16bf;
typedef __attribute__((ext_vector_type(8)))  __bf16 v8bf;
typedef __attribute__((ext_vector_type(8)))  float  v8f;

#define TM  64    // Cout tile per workgroup
#define TN  128   // output-position tile per workgroup
#define BK  32    // WMMA K depth (bf16)
#define BKS 64    // K staged per LDS buffer (2 WMMA steps per barrier)

// ---------------------------------------------------------------------------
// Weight pre-pack: fp32 OIHW -> bf16 [CoutPad][Kpad] (Kpad % 64 == 0), zeros.
// GEMM-K is already contiguous in OIHW: w[oc*Ktot + k].
// ---------------------------------------------------------------------------
__global__ void pack_w_kernel(const float* __restrict__ w, bf16_t* __restrict__ wp,
                              int Cout, int Ktot, int Kpad, int total)
{
    const int i = blockIdx.x * blockDim.x + threadIdx.x;
    if (i >= total) return;
    const int oc = (unsigned)i / (unsigned)Kpad;
    const int k  = i - oc * Kpad;
    const float v = (oc < Cout && k < Ktot) ? w[(size_t)oc * Ktot + k] : 0.f;
    wp[i] = (bf16_t)v;
}

// ---------------------------------------------------------------------------
// Implicit-GEMM conv, templated on the window so k->(ci,ky,kx) uses constant
// divisors (vanishes for 1x1).
//
//   GEMM: M=Cout (tile 64), N=B*Ho*Wo (tile 128), K=Cin*KH*KW
//   8 waves in 4x2: (row-tile wm 0..3) x (col-half wn 0..1)
//   LDS holds only B (im2col fp32->bf16), double-buffered 64-deep:
//   one barrier per 2 WMMA K-steps; per iteration per wave:
//   2 A frags (aligned v8bf global loads of packed L2-resident weights),
//   8 B frags from LDS, 8 x v_wmma_f32_16x16x32_bf16 on 4 acc chains.
//
// Fragment layouts used here (contiguous k, not the ISA table order):
//   A: lane<16 row=lane, k0..15 contiguous; lane>=16 row=lane-16, k16..31
//   B: lane<16 col=lane, k0..15 contiguous; lane>=16 col=lane-16, k16..31
// ---------------------------------------------------------------------------
template<int KH, int KW>
__global__ __launch_bounds__(256)
void conv_wmma_bf16(const bf16_t* __restrict__ wp, const float* __restrict__ x,
                    const float* __restrict__ scale, const float* __restrict__ bias,
                    const float* __restrict__ res, float* __restrict__ y,
                    int N, int Cin, int H, int W, int Cout, int Ho, int Wo,
                    int stride, int pad, int relu, int Kpad)
{
    constexpr int KHW = KH * KW;
    const int Ktot  = Cin * KHW;
    const int Ngemm = N * Ho * Wo;

    __shared__ __align__(32) bf16_t Bs[2][TN * BKS];   // 2 x 16 KB

    const int tid  = threadIdx.x;
    const int lane = tid & 31;
    const int wid  = tid >> 5;
    const int wm   = wid & 3;    // row tile (16 rows)
    const int wn   = wid >> 2;   // col half (64 cols)
    const int m0   = blockIdx.y * TM;
    const int n0   = blockIdx.x * TN;

    // ---- per-thread im2col state: 4 columns x 8-deep k segment ----
    const int colpart = tid >> 3;        // 0..31
    const int seg     = (tid & 7) * 8;   // 0..56
    const float* xp[4];
    int ihb[4], iwb[4];
    bool cok[4];
    #pragma unroll
    for (int s2 = 0; s2 < 4; ++s2) {
        const int n = n0 + colpart + s2 * 32;
        cok[s2] = (n < Ngemm);
        unsigned b = 0, oh = 0, ow = 0;
        if (cok[s2]) {
            const unsigned hw = (unsigned)(Ho * Wo);
            b = (unsigned)n / hw;
            const unsigned r2 = (unsigned)n - b * hw;
            oh = r2 / (unsigned)Wo;
            ow = r2 - oh * (unsigned)Wo;
        }
        xp[s2]  = x + (size_t)b * Cin * H * W;
        ihb[s2] = (int)oh * stride - pad;
        iwb[s2] = (int)ow * stride - pad;
    }

    auto stageB = [&](int buf, int k0) {
        #pragma unroll
        for (int s2 = 0; s2 < 4; ++s2) {
            const int col = colpart + s2 * 32;
            bf16_t* dst = &Bs[buf][col * BKS + seg];
            #pragma unroll
            for (int j = 0; j < 8; ++j) {
                const int k = k0 + seg + j;
                float v = 0.f;
                if (cok[s2] && k < Ktot) {
                    const int ci = k / KHW;            // constant divisor
                    const int rr = k - ci * KHW;
                    const int ky = rr / KW;            // constant divisor
                    const int kx = rr - ky * KW;
                    const int ih = ihb[s2] + ky;
                    const int iw = iwb[s2] + kx;
                    if ((unsigned)ih < (unsigned)H && (unsigned)iw < (unsigned)W)
                        v = xp[s2][((size_t)ci * H + ih) * W + iw];
                }
                dst[j] = (bf16_t)v;
            }
        }
    };

    // ---- A fragment source: packed bf16 weights, direct from global/L2 ----
    const int rlo = lane & 15;
    const int kbA = (lane < 16) ? 0 : 16;
    const int kbB = (lane < 16) ? 0 : 16;
    const bf16_t* arow = wp + (size_t)(m0 + wm * 16 + rlo) * Kpad;

    v8f acc[4];
    #pragma unroll
    for (int t = 0; t < 4; ++t) acc[t] = (v8f){0.f,0.f,0.f,0.f,0.f,0.f,0.f,0.f};

    stageB(0, 0);
    int buf = 0;
    for (int k0 = 0; k0 < Ktot; k0 += BKS) {
        __syncthreads();
        if (k0 + BKS < Ktot) stageB(buf ^ 1, k0 + BKS);

        #pragma unroll
        for (int h = 0; h < 2; ++h) {
            const int kh0 = k0 + h * BK;   // Kpad%64==0 -> A loads stay in pad
            v8bf alo = *(const v8bf*)(arow + kh0 + kbA);
            v8bf ahi = *(const v8bf*)(arow + kh0 + kbA + 8);
            v16bf afrag = __builtin_shufflevector(alo, ahi,
                             0,1,2,3,4,5,6,7,8,9,10,11,12,13,14,15);
            #pragma unroll
            for (int t = 0; t < 4; ++t) {
                v16bf bfrag = *(const v16bf*)(
                    &Bs[buf][(wn * 64 + t * 16 + rlo) * BKS + h * BK + kbB]);
                acc[t] = __builtin_amdgcn_wmma_f32_16x16x32_bf16(
                             false, afrag, false, bfrag, (short)0, acc[t], false, false);
            }
        }
        buf ^= 1;
    }

    // ---- epilogue: folded-BN scale/bias, residual, ReLU, NCHW scatter ----
    const int rowadd = (lane < 16) ? 0 : 8;
    #pragma unroll
    for (int t = 0; t < 4; ++t) {
        const int n = n0 + wn * 64 + t * 16 + rlo;
        if (n >= Ngemm) continue;
        const unsigned hw = (unsigned)(Ho * Wo);
        const unsigned b2 = (unsigned)n / hw;
        const unsigned r2 = (unsigned)n - b2 * hw;
        const unsigned o_h = r2 / (unsigned)Wo;
        const unsigned o_w = r2 - o_h * (unsigned)Wo;
        #pragma unroll
        for (int rr = 0; rr < 8; ++rr) {
            const int oc = m0 + wm * 16 + rr + rowadd;
            if (oc >= Cout) continue;
            float v = acc[t][rr];
            const float s  = scale ? scale[oc] : 1.f;
            const float b3 = bias  ? bias[oc]  : 0.f;
            v = v * s + b3;
            const size_t oidx = (((size_t)b2 * Cout + oc) * Ho + o_h) * Wo + o_w;
            if (res)  v += res[oidx];
            if (relu) v = v > 0.f ? v : 0.f;
            y[oidx] = v;
        }
    }
}

// ---------------------------------------------------------------------------
// Fold frozen BatchNorm into per-channel scale/bias.
// ---------------------------------------------------------------------------
__global__ void fold_bn_kernel(const float* __restrict__ g, const float* __restrict__ b,
                               const float* __restrict__ m, const float* __restrict__ v,
                               float* __restrict__ sc, float* __restrict__ bi, int C)
{
    const int i = blockIdx.x * blockDim.x + threadIdx.x;
    if (i < C) {
        const float s = g[i] * rsqrtf(v[i] + 1e-5f);
        sc[i] = s;
        bi[i] = b[i] - m[i] * s;
    }
}

// ---------------------------------------------------------------------------
// 3x3 stride-2 pad-1 max pool (NCHW).
// ---------------------------------------------------------------------------
__global__ void maxpool3x3s2_kernel(const float* __restrict__ x, float* __restrict__ y,
                                    int N, int C, int H, int W, int Ho, int Wo)
{
    const int i = blockIdx.x * blockDim.x + threadIdx.x;
    const int total = N * C * Ho * Wo;
    if (i >= total) return;
    const unsigned ow = (unsigned)i % (unsigned)Wo;
    const unsigned oh = ((unsigned)i / (unsigned)Wo) % (unsigned)Ho;
    const unsigned c  = ((unsigned)i / (unsigned)(Wo * Ho)) % (unsigned)C;
    const unsigned n  = (unsigned)i / (unsigned)(Wo * Ho * C);
    float best = -INFINITY;
    #pragma unroll
    for (int dy = 0; dy < 3; ++dy) {
        #pragma unroll
        for (int dx = 0; dx < 3; ++dx) {
            const int ih = (int)oh * 2 - 1 + dy;
            const int iw = (int)ow * 2 - 1 + dx;
            if (ih >= 0 && ih < H && iw >= 0 && iw < W) {
                const float v = x[(((size_t)n * C + c) * H + ih) * W + iw];
                best = v > best ? v : best;
            }
        }
    }
    y[i] = best;
}

// ---------------------------------------------------------------------------
// y[NCHW @ Ho,Wo] += nearest-2x-upsample(p[NCHW @ Ho/2, Wo/2])
// ---------------------------------------------------------------------------
__global__ void upsample2x_add_kernel(float* __restrict__ y, const float* __restrict__ p,
                                      int N, int C, int Ho, int Wo)
{
    const int i = blockIdx.x * blockDim.x + threadIdx.x;
    const int total = N * C * Ho * Wo;
    if (i >= total) return;
    const unsigned ow = (unsigned)i % (unsigned)Wo;
    const unsigned oh = ((unsigned)i / (unsigned)Wo) % (unsigned)Ho;
    const unsigned c  = ((unsigned)i / (unsigned)(Wo * Ho)) % (unsigned)C;
    const unsigned n  = (unsigned)i / (unsigned)(Wo * Ho * C);
    const unsigned Hs = Ho >> 1, Ws = Wo >> 1;
    y[i] += p[(((size_t)n * C + c) * Hs + (oh >> 1)) * Ws + (ow >> 1)];
}

// ---------------------------------------------------------------------------
// Assemble final output slice for one pyramid level:
//   out[b, base + (h*W+w)*9 + a, 0:4] = reg[b, a*4+j, h, w]
//   out[b, base + (h*W+w)*9 + a, 4+c] = sigmoid(cls[b, a*80+c, h, w])
// ---------------------------------------------------------------------------
__global__ void assemble_kernel(const float* __restrict__ reg, const float* __restrict__ cls,
                                float* __restrict__ out, int Hf, int base, int Atotal)
{
    const int per   = Hf * Hf * 9;
    const int total = 4 * per;
    const int i = blockIdx.x * blockDim.x + threadIdx.x;
    if (i >= total) return;
    const unsigned b = (unsigned)i / (unsigned)per;
    const unsigned r = (unsigned)i - b * per;
    const unsigned cell = r / 9u;
    const unsigned a = r - cell * 9u;
    const unsigned h = cell / (unsigned)Hf;
    const unsigned w = cell - h * (unsigned)Hf;
    float* o = out + ((size_t)b * Atotal + base + r) * 84;
    #pragma unroll
    for (int j = 0; j < 4; ++j)
        o[j] = reg[(((size_t)b * 36 + a * 4 + j) * Hf + h) * Hf + w];
    for (int c = 0; c < 80; ++c) {
        const float v = cls[(((size_t)b * 720 + a * 80 + c) * Hf + h) * Hf + w];
        o[4 + c] = 1.f / (1.f + __expf(-v));
    }
}

// ---------------------------------------------------------------------------
// Host orchestration.
// Input order = JAX pytree flatten of {'img_batch', 'params'} with dict keys
// sorted alphabetically at every level (bn dicts: b,g,m,v).
// ---------------------------------------------------------------------------
extern "C" void kernel_launch(void* const* d_in, const int* in_sizes, int n_in,
                              void* d_out, int out_size, void* d_ws, size_t ws_size,
                              hipStream_t stream)
{
    (void)in_sizes; (void)n_in; (void)out_size; (void)ws_size;
    auto IN = [&](int i) -> const float* { return (const float*)d_in[i]; };
    float* out = (float*)d_out;

    // ---- workspace carve-out (floats; every slice multiple of 1024) ----
    size_t off = 0;
    auto alloc = [&](size_t n) { float* p = (float*)d_ws + off; off += n; return p; };
    float*  G0    = alloc(16777216);  // stem output 4x64x256x256
    float*  bufs[4];
    for (int i = 0; i < 4; ++i) bufs[i] = alloc(4194304);
    float*  C4buf = alloc(1048576);   // 4x256x32x32
    float*  C5buf = alloc(524288);    // 4x512x16x16
    float*  P4o   = alloc(1048576);
    float*  P5o   = alloc(262144);
    float*  P6o   = alloc(25600);
    bf16_t* wpack = (bf16_t*)alloc(2097152);  // 4M bf16 (max packed: 256x12800)
    float*  sc    = alloc(1024);
    float*  bi    = alloc(1024);

    auto fold = [&](int bnIdx, int C) {
        fold_bn_kernel<<<(C + 255) / 256, 256, 0, stream>>>(
            IN(bnIdx + 1), IN(bnIdx + 0), IN(bnIdx + 2), IN(bnIdx + 3), sc, bi, C);
    };
    auto conv = [&](const float* x, const float* w, const float* scal, const float* bias,
                    const float* res, float* y, int N, int Cin, int H, int W, int Cout,
                    int KH, int KW, int s, int p, int relu) {
        const int Ktot    = Cin * KH * KW;
        const int Kpad    = (Ktot + 63) & ~63;      // %64 for 2-step A loads
        const int CoutPad = (Cout + 63) & ~63;
        const int ptotal  = CoutPad * Kpad;
        pack_w_kernel<<<(ptotal + 255) / 256, 256, 0, stream>>>(w, wpack, Cout, Ktot, Kpad, ptotal);
        const int Ho = (H + 2 * p - KH) / s + 1;
        const int Wo = (W + 2 * p - KW) / s + 1;
        const int Ng = N * Ho * Wo;
        dim3 grid((Ng + TN - 1) / TN, CoutPad / TM);
        switch (KH) {
        case 1: conv_wmma_bf16<1,1><<<grid, 256, 0, stream>>>(wpack, x, scal, bias, res, y,
                    N, Cin, H, W, Cout, Ho, Wo, s, p, relu, Kpad); break;
        case 3: conv_wmma_bf16<3,3><<<grid, 256, 0, stream>>>(wpack, x, scal, bias, res, y,
                    N, Cin, H, W, Cout, Ho, Wo, s, p, relu, Kpad); break;
        case 5: conv_wmma_bf16<5,5><<<grid, 256, 0, stream>>>(wpack, x, scal, bias, res, y,
                    N, Cin, H, W, Cout, Ho, Wo, s, p, relu, Kpad); break;
        default: conv_wmma_bf16<7,7><<<grid, 256, 0, stream>>>(wpack, x, scal, bias, res, y,
                    N, Cin, H, W, Cout, Ho, Wo, s, p, relu, Kpad); break;
        }
    };

    // ================= stem =================
    // indices: stem.bn1 @126 (b,g,m,v), stem.conv1_w @130
    fold(126, 64);
    conv(IN(0), IN(130), sc, bi, nullptr, G0, 4, 3, 512, 512, 64, 7, 7, 2, 3, 1);
    {
        const int total = 4 * 64 * 128 * 128;
        maxpool3x3s2_kernel<<<(total + 255) / 256, 256, 0, stream>>>(
            G0, bufs[0], 4, 64, 256, 256, 128, 128);
    }

    // ================= ResNet-18 blocks =================
    struct Blk { int cin, cout, hin, s, bn1, bn2, c1, c2, dbn, dw; };
    const Blk blks[8] = {
        { 64,  64, 128, 1,  21,  25,  29,  30,  -1,  -1},
        { 64,  64, 128, 1,  31,  35,  39,  40,  -1,  -1},
        { 64, 128, 128, 2,  41,  45,  49,  50,  51,  55},
        {128, 128,  64, 1,  56,  60,  64,  65,  -1,  -1},
        {128, 256,  64, 2,  66,  70,  74,  75,  76,  80},
        {256, 256,  32, 1,  81,  85,  89,  90,  -1,  -1},
        {256, 512,  32, 2,  91,  95,  99, 100, 101, 105},
        {512, 512,  16, 1, 106, 110, 114, 115,  -1,  -1},
    };
    int xi = 0;
    for (int bix = 0; bix < 8; ++bix) {
        const Blk& B = blks[bix];
        const int hout = (B.hin - 1) / B.s + 1;
        float* X = bufs[xi];
        float* D = bufs[(xi + 1) & 3];
        float* T = bufs[(xi + 2) & 3];
        float* Y = bufs[(xi + 3) & 3];
        const float* R = X;
        if (B.dw >= 0) {
            fold(B.dbn, B.cout);
            conv(X, IN(B.dw), sc, bi, nullptr, D, 4, B.cin, B.hin, B.hin, B.cout,
                 1, 1, B.s, 0, 0);
            R = D;
        }
        fold(B.bn1, B.cout);
        conv(X, IN(B.c1), sc, bi, nullptr, T, 4, B.cin, B.hin, B.hin, B.cout,
             3, 3, B.s, 1, 1);
        fold(B.bn2, B.cout);
        conv(T, IN(B.c2), sc, bi, R, Y, 4, B.cout, hout, hout, B.cout,
             3, 3, 1, 1, 1);
        xi = (xi + 3) & 3;
        if (bix == 5)   // C4 (4,256,32,32)
            hipMemcpyAsync(C4buf, Y, sizeof(float) * 4 * 256 * 32 * 32,
                           hipMemcpyDeviceToDevice, stream);
        if (bix == 7)   // C5 (4,512,16,16)
            hipMemcpyAsync(C5buf, Y, sizeof(float) * 4 * 512 * 16 * 16,
                           hipMemcpyDeviceToDevice, stream);
    }

    // ================= FPN =================
    // fpn indices: P4_1_b=11 P4_1_w=12 P4_2_b=13 P4_2_w=14
    //              P5_1_b=15 P5_1_w=16 P5_2_b=17 P5_2_w=18 P6_b=19 P6_w=20
    float* P5buf = bufs[0];
    float* P4buf = bufs[1];
    conv(C5buf, IN(16), nullptr, IN(15), nullptr, P5buf, 4, 512, 16, 16, 256, 1, 1, 1, 0, 0);
    conv(P5buf, IN(18), nullptr, IN(17), nullptr, P5o,   4, 256, 16, 16, 256, 3, 3, 1, 1, 0);
    conv(C4buf, IN(12), nullptr, IN(11), nullptr, P4buf, 4, 256, 32, 32, 256, 1, 1, 1, 0, 0);
    {
        const int total = 4 * 256 * 32 * 32;
        upsample2x_add_kernel<<<(total + 255) / 256, 256, 0, stream>>>(
            P4buf, P5buf, 4, 256, 32, 32);
    }
    conv(P4buf, IN(14), nullptr, IN(13), nullptr, P4o, 4, 256, 32, 32, 256, 3, 3, 1, 1, 0);
    conv(C5buf, IN(20), nullptr, IN(19), nullptr, P6o, 4, 512, 16, 16, 256, 5, 5, 3, 1, 0);

    // ================= heads + assembly =================
    // cls: convs[0].b=1 ... out_b=9 out_w=10 ; reg: convs[0].b=116 ... out_b=124 out_w=125
    auto run_head = [&](const float* ft, int Hf, int cbase, int out_b, int out_w,
                        int outC, float* result) {
        const float* h = ft;
        float* ping = bufs[0];
        float* pong = bufs[1];
        for (int i = 0; i < 4; ++i) {
            conv(h, IN(cbase + 2 * i + 1), nullptr, IN(cbase + 2 * i), nullptr, ping,
                 4, 256, Hf, Hf, 256, 3, 3, 1, 1, 1);
            h = ping;
            float* t = ping; ping = pong; pong = t;
        }
        conv(h, IN(out_w), nullptr, IN(out_b), nullptr, result,
             4, 256, Hf, Hf, outC, 3, 3, 1, 1, 0);
    };

    const float* feats[3] = {P4o, P5o, P6o};
    const int    Hfs[3]   = {32, 16, 5};
    const int    bases[3] = {0, 9216, 11520};
    const int    Atotal   = 11745;
    float* regbuf = bufs[2];
    float* clsbuf = bufs[3];
    for (int f = 0; f < 3; ++f) {
        run_head(feats[f], Hfs[f], 116, 124, 125, 36,  regbuf);
        run_head(feats[f], Hfs[f],   1,   9,  10, 720, clsbuf);
        const int total = 4 * Hfs[f] * Hfs[f] * 9;
        assemble_kernel<<<(total + 255) / 256, 256, 0, stream>>>(
            regbuf, clsbuf, out, Hfs[f], bases[f], Atotal);
    }
}
